// JaggedFlashAttnBlock_67164698575450
// MI455X (gfx1250) — matrix operand order repaired and checked by reading, not compiled, not run on hardware
//
#include <hip/hip_runtime.h>
#include <hip/hip_bf16.h>
#include <math.h>

typedef _Float16 f16;
typedef __attribute__((ext_vector_type(16))) _Float16 v16h;
typedef __attribute__((ext_vector_type(8)))  _Float16 v8h;
typedef __attribute__((ext_vector_type(4)))  _Float16 v4h;
typedef __attribute__((ext_vector_type(8)))  float    v8f;
typedef __attribute__((ext_vector_type(4)))  float    v4f;
typedef __attribute__((ext_vector_type(4)))  unsigned int v4u;
typedef __attribute__((ext_vector_type(8)))  unsigned int v8u;

#define LNUM   4
#define BDIM   2
#define SDIM   2048
#define HDIM   1024
#define NHEAD  16
#define HDHEAD 64
#define FFNDIM 4096
#define MROWS  (BDIM*SDIM)   // 4096 token rows

#define CAT16(lo,hi) __builtin_shufflevector(lo,hi,0,1,2,3,4,5,6,7,8,9,10,11,12,13,14,15)

static __device__ __forceinline__ v8f v8f_zero() {
  v8f z = {0.f,0.f,0.f,0.f,0.f,0.f,0.f,0.f};
  return z;
}

static __device__ __forceinline__ v8f wmma16(v16h a, v16h b, v8f c) {
  // D = A(16x32 f16) * B(32x16 f16) + C(16x16 f32)
  return __builtin_amdgcn_wmma_f32_16x16x32_f16(false, a, false, b, (short)0, c, false, false);
}

// ---- CDNA5 async global -> LDS copy (ASYNCcnt path, no VGPR round-trip) ----
static __device__ __forceinline__ void async_cp16(const f16* gptr, const f16* lptr) {
  // generic LDS address = aperture_hi32 | lds_offset_lo32 -> truncate for vdst
  unsigned lds = (unsigned)(uintptr_t)lptr;
  unsigned long long ga = (unsigned long long)(uintptr_t)gptr;
  asm volatile("global_load_async_to_lds_b128 %0, %1, off"
               :: "v"(lds), "v"(ga) : "memory");
}
static __device__ __forceinline__ void wait_async0() {
  asm volatile("s_wait_asynccnt 0" ::: "memory");
}

// ---- CDNA5 Tensor Data Mover: DMA a 2D tile (row-major) into LDS ----------
// tile_d0 = elements per row (f16), tile_d1 = rows, stride = row stride in
// elements.  LDS padding: after (1<<(pi+1)) DWORDs stored, skip (pa+1) DWORDs.
static __device__ __forceinline__ void tdm_load_2d(const f16* gaddr,
                                                   const f16* laddr,
                                                   unsigned tile_d0,
                                                   unsigned tile_d1,
                                                   unsigned stride,
                                                   unsigned pi, unsigned pa)
{
  unsigned long long ga = (unsigned long long)(uintptr_t)gaddr;
  unsigned lds = (unsigned)(uintptr_t)laddr;
  v4u g0;
  g0[0] = 1u;                                                 // count=1, user D#
  g0[1] = lds;                                                // lds_addr (bytes)
  g0[2] = (unsigned)(ga & 0xFFFFFFFFu);                       // global_addr[31:0]
  g0[3] = (unsigned)((ga >> 32) & 0x01FFFFFFu) | (2u << 30);  // addr[56:32]|type=2
  v8u g1;
  g1[0] = (1u << 16)            // data_size = 2 bytes
        | (1u << 20)            // pad_enable
        | (pi << 22)            // pad_interval code
        | (pa << 25);           // pad_amount code
  g1[1] = tile_d0 << 16;        // tensor_dim0[15:0] (== tile -> never OOB)
  g1[2] = tile_d1 << 16;        // tensor_dim0 hi | tensor_dim1[15:0]
  g1[3] = tile_d0 << 16;        // tensor_dim1 hi | tile_dim0
  g1[4] = tile_d1;              // tile_dim1 | tile_dim2=0
  g1[5] = stride;               // tensor_dim0_stride[31:0] (elements)
  g1[6] = 0u;
  g1[7] = 0u;
  asm volatile("tensor_load_to_lds %0, %1" :: "s"(g0), "s"(g1) : "memory");
}

// ---------------- fp32 -> fp16 weight conversion (x4 vectorized) -------------
__global__ __launch_bounds__(256) void cvt_kernel(const float* __restrict__ s,
                                                  f16* __restrict__ d, int n4) {
  int i = blockIdx.x * 256 + threadIdx.x;
  if (i < n4) {
    v4f x = *(const v4f*)(s + (size_t)i * 4);
    *(v4h*)(d + (size_t)i * 4) = __builtin_convertvector(x, v4h);
  }
}

// ---------------- layernorm (one block per row of H=1024) ----------------
template<int OUT16>
__global__ __launch_bounds__(256) void ln_kernel(const float* __restrict__ x,
                                                 const float* __restrict__ g,
                                                 const float* __restrict__ b,
                                                 f16* __restrict__ o16,
                                                 float* __restrict__ o32) {
  int row = blockIdx.x;
  const float* xr = x + (size_t)row * HDIM;
  int t = threadIdx.x;
  float v[4], s = 0.f, s2 = 0.f;
#pragma unroll
  for (int i = 0; i < 4; ++i) { v[i] = xr[t + 256*i]; s += v[i]; s2 += v[i]*v[i]; }
#pragma unroll
  for (int m = 16; m >= 1; m >>= 1) { s += __shfl_xor(s, m, 32); s2 += __shfl_xor(s2, m, 32); }
  __shared__ float rs[8], rq[8];
  int w = t >> 5, lane = t & 31;
  if (lane == 0) { rs[w] = s; rq[w] = s2; }
  __syncthreads();
  if (t < 32) {
    float a  = (lane < 8) ? rs[lane] : 0.f;
    float a2 = (lane < 8) ? rq[lane] : 0.f;
#pragma unroll
    for (int m = 4; m >= 1; m >>= 1) { a += __shfl_xor(a, m, 32); a2 += __shfl_xor(a2, m, 32); }
    if (lane == 0) { rs[0] = a; rq[0] = a2; }
  }
  __syncthreads();
  float mu   = rs[0] * (1.f / HDIM);
  float var  = rq[0] * (1.f / HDIM) - mu * mu;
  float rstd = rsqrtf(var + 1e-5f);
#pragma unroll
  for (int i = 0; i < 4; ++i) {
    int c = t + 256*i;
    float y = (v[i] - mu) * rstd * g[c] + b[c];
    if (OUT16) o16[(size_t)row*HDIM + c] = (f16)y;
    else       o32[(size_t)row*HDIM + c] = y;
  }
}

// ---------------- tiled WMMA GEMM: C(MxN) = A(MxK) * B(KxN), f16 in, f32 acc --
// TDM (tensor_load_to_lds) double-buffered pipeline: wave 0 DMAs both tiles,
// compute waves only touch LDS fragments + WMMA.
// MT = M-tile (128 or 256).  MT=256: 4x2 wave grid, 64x64/wave, 16 WMMA/K-step.
//                            MT=128: 2x4 wave grid, 64x32/wave,  8 WMMA/K-step.
// EPI: 0 = scatter into q/kT/v (f16), 1/3 = residual add into f32 buf,
//      2 = exact GELU -> f16
template<int EPI, int MT>
__global__ __launch_bounds__(256) void gemm_kernel(
    const f16* __restrict__ A, const f16* __restrict__ Bw, int N, int K,
    float* __restrict__ xres, f16* __restrict__ oq, f16* __restrict__ okT,
    f16* __restrict__ ov, f16* __restrict__ o16)
{
  constexpr int TN = (MT == 256) ? 4 : 2;       // n-tiles per wave
  __shared__ alignas(16) f16 sA[2][MT][40];     // MT x 32 (+8 pad = 4 DW)
  __shared__ alignas(16) f16 sB[2][32][136];    // 32 x 128 (+8 pad = 4 DW)
  int t = threadIdx.x;
  int lane = t & 31, wid = t >> 5;
  int wm = (MT == 256) ? (wid >> 1) : (wid >> 2);
  int wn = (MT == 256) ? (wid & 1)  : (wid & 3);
  int lh = lane >> 4, ln = lane & 15, loff = lh * 8;
  int mb = blockIdx.y * MT, nb = blockIdx.x * 128;
  bool tdm_wave = (wid == 0);

  v8f acc[4][TN];
#pragma unroll
  for (int i = 0; i < 4; ++i)
#pragma unroll
    for (int j = 0; j < TN; ++j) acc[i][j] = v8f_zero();

  if (tdm_wave) {
    // A: MT rows x 32 elems, row = 16 DW -> pad code 3; pad 4 DW -> code 3
    tdm_load_2d(A + (size_t)mb * K, &sA[0][0][0], 32, MT, (unsigned)K, 3, 3);
    // B: 32 rows x 128 elems, row = 64 DW -> pad code 5; pad 4 DW -> code 3
    tdm_load_2d(Bw + nb,            &sB[0][0][0], 128, 32, (unsigned)N, 5, 3);
  }

  for (int kb = 0; kb < K; kb += 32) {
    int p = (kb >> 5) & 1;
    __builtin_amdgcn_s_wait_tensorcnt(0);
    __syncthreads();
    if (kb + 32 < K && tdm_wave) {
      tdm_load_2d(A + (size_t)mb * K + (kb + 32), &sA[p ^ 1][0][0],
                  32, MT, (unsigned)K, 3, 3);
      tdm_load_2d(Bw + (size_t)(kb + 32) * N + nb, &sB[p ^ 1][0][0],
                  128, 32, (unsigned)N, 5, 3);
    }

    v16h bf[TN];
#pragma unroll
    for (int nt = 0; nt < TN; ++nt) {
#pragma unroll
      for (int e = 0; e < 16; ++e)
        bf[nt][e] = sB[p][loff + (e & 7) + ((e >> 3) << 4)][wn*(TN*16) + nt*16 + ln];
    }
#pragma unroll
    for (int mt = 0; mt < 4; ++mt) {
      int r = wm*64 + mt*16 + ln;
      v8h lo = *(const v8h*)&sA[p][r][loff];
      v8h hi = *(const v8h*)&sA[p][r][16 + loff];
      v16h af = CAT16(lo, hi);
#pragma unroll
      for (int nt = 0; nt < TN; ++nt)
        acc[mt][nt] = wmma16(af, bf[nt], acc[mt][nt]);
    }
    // no trailing barrier: LDS reads are consumed by WMMA (DS-waited) before
    // any wave can pass the next iteration's barrier and let the TDM overwrite
  }

  // epilogue (C layout: row = v + 8*(lane>=16), col = lane%16)
#pragma unroll
  for (int mt = 0; mt < 4; ++mt)
#pragma unroll
    for (int nt = 0; nt < TN; ++nt)
#pragma unroll
      for (int e = 0; e < 8; ++e) {
        int row = mb + wm*64 + mt*16 + e + 8*lh;
        int col = nb + wn*(TN*16) + nt*16 + ln;
        float val = acc[mt][nt][e];
        if (EPI == 0) {
          int which = col >> 10;            // 0=q 1=k 2=v
          int hq    = (col >> 6) & 15;      // head
          int hd    = col & 63;
          int bb2   = row >> 11;            // batch
          int s     = row & 2047;           // seq
          if (which == 0)
            oq [(((size_t)bb2*NHEAD + hq)*SDIM + s)*HDHEAD + hd] = (f16)val;
          else if (which == 1)
            okT[(((size_t)bb2*NHEAD + hq)*HDHEAD + hd)*SDIM + s] = (f16)val;   // K^T
          else
            ov [(((size_t)bb2*NHEAD + hq)*SDIM + s)*HDHEAD + hd] = (f16)val;
        } else if (EPI == 1 || EPI == 3) {
          size_t idx = (size_t)row * N + col;
          xres[idx] = xres[idx] + val;
        } else if (EPI == 2) {
          o16[(size_t)row * N + col] =
            (f16)(0.5f * val * (1.f + erff(val * 0.70710678118654752f)));
        }
      }
}

// ---------------- flash attention: one block per (b, head, 128 q-rows) -------
__global__ __launch_bounds__(256) void attn_kernel(const f16* __restrict__ q,
                                                   const f16* __restrict__ kT,
                                                   const f16* __restrict__ vsrc,
                                                   f16* __restrict__ o)
{
  __shared__ alignas(16) f16 sKT[2][64][72];    // [hd][key]   (64-key block)
  __shared__ alignas(16) f16 sV [2][64][72];    // [key][hd]
  __shared__ alignas(16) f16 sP [8][16][72];    // per-wave probs staging

  int t = threadIdx.x;
  int lane = t & 31, wq = t >> 5;
  int lh = lane >> 4, ln = lane & 15, loff = lh * 8;
  int qb = blockIdx.x, hh = blockIdx.y, bb = blockIdx.z;
  size_t bh = (size_t)(bb * NHEAD + hh);
  const f16* qg = q    + bh * SDIM * HDHEAD;
  const f16* kg = kT   + bh * HDHEAD * SDIM;
  const f16* vg = vsrc + bh * SDIM * HDHEAD;

  // per-thread chunk mapping for K^T / V tile copies (64 x 64 f16 each)
  int r0 = t >> 3,         of0 = (t & 7) * 8;
  int r1 = (t + 256) >> 3, of1 = ((t + 256) & 7) * 8;

  auto issue_kv = [&](int j, int p) {
    async_cp16(kg + (size_t)r0 * SDIM + j*64 + of0, &sKT[p][r0][of0]);
    async_cp16(kg + (size_t)r1 * SDIM + j*64 + of1, &sKT[p][r1][of1]);
    async_cp16(vg + (size_t)(j*64 + r0) * HDHEAD + of0, &sV[p][r0][of0]);
    async_cp16(vg + (size_t)(j*64 + r1) * HDHEAD + of1, &sV[p][r1][of1]);
  };

  // preload Q fragments for this wave's 16 rows (A layout, HD=64 = 2 k-chunks);
  // fold in the 1/sqrt(64)=0.125 softmax scale (exact in f16)
  int qrow = qb*128 + wq*16 + ln;
  v16h aq[2];
#pragma unroll
  for (int c = 0; c < 2; ++c) {
    const f16* p = qg + (size_t)qrow * HDHEAD + c*32 + loff;
    v8h lo = *(const v8h*)p;
    v8h hi = *(const v8h*)(p + 16);
    v16h a = CAT16(lo, hi);
#pragma unroll
    for (int e = 0; e < 16; ++e) a[e] = a[e] * (f16)0.125f;
    aq[c] = a;
  }

  // "ones column" B fragment: B[k][0]=1, else 0 -> P@bones gives row sums,
  // maintained in the O5 accumulator by the same corr-rescale recurrence.
  v16h bones;
#pragma unroll
  for (int e = 0; e < 16; ++e) bones[e] = (ln == 0) ? (f16)1.f : (f16)0.f;

  float mrow[8];
  v8f Oacc[4], O5;
#pragma unroll
  for (int e = 0; e < 8; ++e) mrow[e] = -INFINITY;
#pragma unroll
  for (int i = 0; i < 4; ++i) Oacc[i] = v8f_zero();
  O5 = v8f_zero();

  int jmax = 2*qb + 1;
  issue_kv(0, 0);
  for (int j = 0; j <= jmax; ++j) {
    int p = j & 1;
    wait_async0();
    __syncthreads();
    if (j < jmax) issue_kv(j + 1, p ^ 1);

    // S = Q * K^T  (4 n-tiles of 16 keys, 2 hd-chunks)
    v8f sc[4];
#pragma unroll
    for (int nt = 0; nt < 4; ++nt) {
      v8f a = v8f_zero();
#pragma unroll
      for (int c = 0; c < 2; ++c) {
        v16h bk;
#pragma unroll
        for (int e = 0; e < 16; ++e)
          bk[e] = sKT[p][c*32 + loff + (e & 7) + ((e >> 3) << 4)][nt*16 + ln];
        a = wmma16(aq[c], bk, a);
      }
      sc[nt] = a;
    }

    bool needMask = (j >= 2*qb);
    if (needMask) {
#pragma unroll
      for (int nt = 0; nt < 4; ++nt)
#pragma unroll
        for (int e = 0; e < 8; ++e) {
          int col = j*64 + nt*16 + ln;
          int row = qb*128 + wq*16 + e + 8*lh;
          if (col > row) sc[nt][e] = -INFINITY;
        }
    }

    // online softmax: row max (cross-lane), correction, probs
    float mnew[8], corr[8];
#pragma unroll
    for (int e = 0; e < 8; ++e) {
      float m = fmaxf(fmaxf(sc[0][e], sc[1][e]), fmaxf(sc[2][e], sc[3][e]));
#pragma unroll
      for (int d = 8; d >= 1; d >>= 1) m = fmaxf(m, __shfl_xor(m, d, 32));
      mnew[e] = fmaxf(mrow[e], m);
      corr[e] = __expf(mrow[e] - mnew[e]);
      mrow[e] = mnew[e];
    }
#pragma unroll
    for (int nt = 0; nt < 4; ++nt)
#pragma unroll
      for (int e = 0; e < 8; ++e) {
        float pv = __expf(sc[nt][e] - mnew[e]);
        sP[wq][e + 8*lh][nt*16 + ln] = (f16)pv;   // C-layout -> LDS
      }
#pragma unroll
    for (int nt = 0; nt < 4; ++nt)
#pragma unroll
      for (int e = 0; e < 8; ++e) Oacc[nt][e] *= corr[e];
#pragma unroll
    for (int e = 0; e < 8; ++e) O5[e] *= corr[e];   // running row-sum rescale

    // O += P * V ; O5 += P * ones  (A-fragments re-read from per-wave LDS tile)
#pragma unroll
    for (int c = 0; c < 2; ++c) {
      v8h lo = *(const v8h*)&sP[wq][ln][c*32 + loff];
      v8h hi = *(const v8h*)&sP[wq][ln][c*32 + 16 + loff];
      v16h ap = CAT16(lo, hi);
#pragma unroll
      for (int nt = 0; nt < 4; ++nt) {
        v16h bv;
#pragma unroll
        for (int e = 0; e < 16; ++e)
          bv[e] = sV[p][c*32 + loff + (e & 7) + ((e >> 3) << 4)][nt*16 + ln];
        Oacc[nt] = wmma16(ap, bv, Oacc[nt]);
      }
      O5 = wmma16(ap, bones, O5);
    }
  }

  // row sums live in column 0 (other columns are exactly 0) -> one final
  // cross-lane broadcast, then normalize and store
  float inv[8];
#pragma unroll
  for (int e = 0; e < 8; ++e) {
    float x = O5[e];
#pragma unroll
    for (int d = 8; d >= 1; d >>= 1) x += __shfl_xor(x, d, 32);
    inv[e] = 1.f / x;
  }
#pragma unroll
  for (int nt = 0; nt < 4; ++nt)
#pragma unroll
    for (int e = 0; e < 8; ++e) {
      int srow = qb*128 + wq*16 + e + 8*lh;
      o[((size_t)bb * SDIM + srow) * HDIM + hh*HDHEAD + nt*16 + ln] =
        (f16)(Oacc[nt][e] * inv[e]);
    }
}

// ---------------- host orchestration ----------------
extern "C" void kernel_launch(void* const* d_in, const int* in_sizes, int n_in,
                              void* d_out, int out_size, void* d_ws, size_t ws_size,
                              hipStream_t stream) {
  (void)in_sizes; (void)n_in; (void)out_size; (void)ws_size;
  const float* hs    = (const float*)d_in[0];
  const float* ln1g  = (const float*)d_in[1];
  const float* ln1b  = (const float*)d_in[2];
  const float* Wqkv  = (const float*)d_in[3];
  const float* Wproj = (const float*)d_in[4];
  const float* ln2g  = (const float*)d_in[5];
  const float* ln2b  = (const float*)d_in[6];
  const float* Wfc1  = (const float*)d_in[7];
  const float* Wfc2  = (const float*)d_in[8];
  const float* lnfg  = (const float*)d_in[9];
  const float* lnfb  = (const float*)d_in[10];

  char* ws = (char*)d_ws;
  size_t off = 0;
  auto take = [&](size_t bytes) -> char* {
    char* p = ws + off;
    off += (bytes + 255) & ~(size_t)255;
    return p;
  };

  f16*   wqkv16  = (f16*)take((size_t)LNUM * HDIM * 3 * HDIM * sizeof(f16));
  f16*   wproj16 = (f16*)take((size_t)LNUM * HDIM * HDIM     * sizeof(f16));
  f16*   wfc1_16 = (f16*)take((size_t)LNUM * HDIM * FFNDIM   * sizeof(f16));
  f16*   wfc2_16 = (f16*)take((size_t)LNUM * FFNDIM * HDIM   * sizeof(f16));
  f16*   h16     = (f16*)take((size_t)MROWS * HDIM           * sizeof(f16));
  f16*   q16     = (f16*)take((size_t)BDIM * NHEAD * SDIM * HDHEAD * sizeof(f16));
  f16*   kT16    = (f16*)take((size_t)BDIM * NHEAD * SDIM * HDHEAD * sizeof(f16));
  f16*   v16p   = (f16*)take((size_t)BDIM * NHEAD * SDIM * HDHEAD * sizeof(f16));
  f16*   att16   = (f16*)take((size_t)MROWS * HDIM           * sizeof(f16));
  f16*   g16     = (f16*)take((size_t)MROWS * FFNDIM         * sizeof(f16));
  float* xbuf    = (float*)take((size_t)MROWS * HDIM         * sizeof(float));

  { int n4 = LNUM*HDIM*3*HDIM/4;  cvt_kernel<<<(n4+255)/256,256,0,stream>>>(Wqkv,  wqkv16,  n4); }
  { int n4 = LNUM*HDIM*HDIM/4;    cvt_kernel<<<(n4+255)/256,256,0,stream>>>(Wproj, wproj16, n4); }
  { int n4 = LNUM*HDIM*FFNDIM/4;  cvt_kernel<<<(n4+255)/256,256,0,stream>>>(Wfc1,  wfc1_16, n4); }
  { int n4 = LNUM*FFNDIM*HDIM/4;  cvt_kernel<<<(n4+255)/256,256,0,stream>>>(Wfc2,  wfc2_16, n4); }

  hipMemcpyAsync(xbuf, hs, (size_t)MROWS * HDIM * sizeof(float),
                 hipMemcpyDeviceToDevice, stream);

  for (int l = 0; l < LNUM; ++l) {
    ln_kernel<1><<<MROWS, 256, 0, stream>>>(xbuf, ln1g + l*HDIM, ln1b + l*HDIM, h16, nullptr);
    // QKV: K=1024, big tile (256x128) -> 24x16 = 384 WGs, 16 WMMA per K-step
    gemm_kernel<0, 256><<<dim3(3*HDIM/128, MROWS/256), 256, 0, stream>>>(
        h16, wqkv16 + (size_t)l*HDIM*3*HDIM, 3*HDIM, HDIM,
        nullptr, q16, kT16, v16p, nullptr);
    attn_kernel<<<dim3(SDIM/128, NHEAD, BDIM), 256, 0, stream>>>(q16, kT16, v16p, att16);
    // proj: N=1024 -> keep 128x128 tile for parallelism (8x32 = 256 WGs)
    gemm_kernel<1, 128><<<dim3(HDIM/128, MROWS/128), 256, 0, stream>>>(
        att16, wproj16 + (size_t)l*HDIM*HDIM, HDIM, HDIM,
        xbuf, nullptr, nullptr, nullptr, nullptr);
    ln_kernel<1><<<MROWS, 256, 0, stream>>>(xbuf, ln2g + l*HDIM, ln2b + l*HDIM, h16, nullptr);
    // fc1: N=4096 -> big tile (256x128): 32x16 = 512 WGs
    gemm_kernel<2, 256><<<dim3(FFNDIM/128, MROWS/256), 256, 0, stream>>>(
        h16, wfc1_16 + (size_t)l*HDIM*FFNDIM, FFNDIM, HDIM,
        nullptr, nullptr, nullptr, nullptr, g16);
    // fc2: N=1024 -> 128x128 tile (8x32 = 256 WGs)
    gemm_kernel<3, 128><<<dim3(HDIM/128, MROWS/128), 256, 0, stream>>>(
        g16, wfc2_16 + (size_t)l*FFNDIM*HDIM, HDIM, FFNDIM,
        xbuf, nullptr, nullptr, nullptr, nullptr);
  }
  ln_kernel<0><<<MROWS, 256, 0, stream>>>(xbuf, lnfg, lnfb, nullptr, (float*)d_out);
}
